// GlobalAndLocalMHA_84473416778160
// MI455X (gfx1250) — matrix operand order repaired and checked by reading, not compiled, not run on hardware
//
#include <hip/hip_runtime.h>

typedef unsigned short u16;
typedef __attribute__((ext_vector_type(16))) __bf16 bf16x16;
typedef __attribute__((ext_vector_type(8)))  float  f32x8;
typedef __attribute__((ext_vector_type(4)))  unsigned uint32x4;
typedef __attribute__((ext_vector_type(8)))  int      int32x8;
typedef __attribute__((ext_vector_type(4)))  int      int32x4;

#if defined(__has_builtin)
#if __has_builtin(__builtin_amdgcn_tensor_load_to_lds)
#define HAVE_TDM 1
#endif
#endif

union Frag { struct { uint4 lo, hi; } u; bf16x16 v; };

static __device__ __forceinline__ u16 f2bf(float f) {
  union { float f; unsigned u; } x; x.f = f;
  unsigned r = x.u + 0x7fffu + ((x.u >> 16) & 1u);
  return (u16)(r >> 16);
}
static __device__ __forceinline__ float bf2f(u16 h) {
  union { unsigned u; float f; } x; x.u = ((unsigned)h) << 16; return x.f;
}
static __device__ __forceinline__ unsigned pack2(u16 a, u16 b) {
  return (unsigned)a | ((unsigned)b << 16);
}
static __device__ __forceinline__ f32x8 wmma_bf16(bf16x16 a, bf16x16 b, f32x8 c) {
  return __builtin_amdgcn_wmma_f32_16x16x32_bf16(false, a, false, b, (short)0, c, false, false);
}

// CDNA5 async copy: global -> LDS, 16 bytes per lane, tracked by ASYNCcnt.
static __device__ __forceinline__ void async_b128(unsigned lds_off, const void* base,
                                                  unsigned byte_off) {
  asm volatile("global_load_async_to_lds_b128 %0, %1, %2"
               :: "v"(lds_off), "v"(byte_off), "s"(base) : "memory");
}
static __device__ __forceinline__ void wait_async_le4() {
  asm volatile("s_wait_asynccnt 0x4" ::: "memory");
}
static __device__ __forceinline__ void wait_async_0() {
  asm volatile("s_wait_asynccnt 0x0" ::: "memory");
}
static __device__ __forceinline__ unsigned lds_addr(const void* p) {
  return (unsigned)(uintptr_t)p;   // generic LDS pointer: addr[31:0] is the LDS offset
}

// ---------------------------------------------------------------------------
// Kernel 0: one-shot conversions so every later tile move is a straight bf16
// copy (async/TDM friendly):  x -> bf16 ;  w_qkv^T, w_out^T -> bf16.
// ---------------------------------------------------------------------------
__global__ void prep_kernel(const float* __restrict__ x, const float* __restrict__ wqkv,
                            const float* __restrict__ wout, u16* __restrict__ xb,
                            u16* __restrict__ wqT, u16* __restrict__ woT) {
  size_t i = (size_t)blockIdx.x * 256 + threadIdx.x;
  if (i < 16777216u) {
    xb[i] = f2bf(x[i]);
  } else if (i < 16777216u + 3145728u) {
    size_t j = i - 16777216u;               // wqT[n][k] = wqkv[k][n]
    size_t n = j >> 10, k = j & 1023;
    wqT[j] = f2bf(wqkv[k * 3072 + n]);
  } else {
    size_t j = i - 16777216u - 3145728u;    // woT[n][k] = wout[k][n]
    size_t n = j >> 10, k = j & 1023;
    woT[j] = f2bf(wout[k * 1024 + n]);
  }
}

// ---------------------------------------------------------------------------
// Kernel 1: qkv = x(16384x1024 bf16) @ w_qkv^T(3072x1024 bf16, B pre-transposed)
// 128x128 tile/block, 8 waves (4Mx2N), double-buffered async LDS staging.
// ---------------------------------------------------------------------------
__global__ __launch_bounds__(256, 1)
void qkv_gemm(const u16* __restrict__ xb, const u16* __restrict__ wqT,
              u16* __restrict__ qb, u16* __restrict__ kb, u16* __restrict__ vb) {
  __shared__ u16 a_lds[2][128 * 40];
  __shared__ u16 b_lds[2][128 * 40];
  const int tid = threadIdx.x, lane = tid & 31, wave = tid >> 5;
  const int lc = lane & 15, kg = lane >> 4;
  const int m0 = blockIdx.y * 128, n0 = blockIdx.x * 128;
  const int wm = wave >> 1, wn = wave & 1;
  f32x8 acc[2][4] = {};

  auto issue = [&](int buf, int kk) {
#pragma unroll
    for (int j = 0; j < 2; ++j) {
      const int idx = tid + j * 256;
      const int row = idx >> 2, pos = idx & 3;
      const unsigned l = (unsigned)((row * 40 + pos * 8) * 2);
      async_b128(lds_addr(&a_lds[buf][0]) + l, xb,
                 (unsigned)(((m0 + row) * 1024 + kk + pos * 8) * 2));
      async_b128(lds_addr(&b_lds[buf][0]) + l, wqT,
                 (unsigned)(((n0 + row) * 1024 + kk + pos * 8) * 2));
    }
  };

  issue(0, 0);
  for (int it = 0; it < 32; ++it) {
    const int buf = it & 1;
    if (it + 1 < 32) { issue(buf ^ 1, (it + 1) * 32); wait_async_le4(); }
    else             { wait_async_0(); }
    __syncthreads();

    Frag a[2], b[4];
#pragma unroll
    for (int mt = 0; mt < 2; ++mt) {
      const int m = wm * 32 + mt * 16 + lc;
      a[mt].u.lo = *(const uint4*)&a_lds[buf][m * 40 + kg * 8];
      a[mt].u.hi = *(const uint4*)&a_lds[buf][m * 40 + kg * 8 + 16];
    }
#pragma unroll
    for (int nt = 0; nt < 4; ++nt) {
      const int nn = wn * 64 + nt * 16 + lc;
      b[nt].u.lo = *(const uint4*)&b_lds[buf][nn * 40 + kg * 8];
      b[nt].u.hi = *(const uint4*)&b_lds[buf][nn * 40 + kg * 8 + 16];
    }
#pragma unroll
    for (int mt = 0; mt < 2; ++mt)
#pragma unroll
      for (int nt = 0; nt < 4; ++nt)
        acc[mt][nt] = wmma_bf16(a[mt].v, b[nt].v, acc[mt][nt]);
    __syncthreads();
  }

#pragma unroll
  for (int mt = 0; mt < 2; ++mt)
#pragma unroll
    for (int nt = 0; nt < 4; ++nt)
#pragma unroll
      for (int r = 0; r < 8; ++r) {
        const int row = m0 + wm * 32 + mt * 16 + r + 8 * kg;
        const int col = n0 + wn * 64 + nt * 16 + lc;
        const int sel = col >> 10, hh = (col >> 6) & 15, d = col & 63;
        const int bb = row >> 13, ns = row & 8191;
        u16* dst = (sel == 0) ? qb : ((sel == 1) ? kb : vb);
        dst[((size_t)(bb * 16 + hh) * 8192 + ns) * 64 + d] = f2bf(acc[mt][nt][r]);
      }
}

// ---------------------------------------------------------------------------
// Kernel 2: per-row l2norm * scale (q also *8) + RoPE (q phase n+128, k phase n)
// ---------------------------------------------------------------------------
__global__ void normrope_kernel(u16* __restrict__ qb, u16* __restrict__ kb,
                                const float* __restrict__ qs, const float* __restrict__ ksc) {
  __shared__ float scq[64], sck[64], invf[32];
  const int tid = threadIdx.x;
  if (tid < 64) scq[tid] = qs[tid];
  else if (tid < 128) sck[tid - 64] = ksc[tid - 64];
  else if (tid < 160) invf[tid - 128] = __expf(-(float)(tid - 128) * (9.210340371976184f / 32.f));
  __syncthreads();

  const size_t r = (size_t)blockIdx.x * 256 + tid;
  const bool isq = r < 262144u;
  const size_t row = isq ? r : (r - 262144u);
  u16* p = (isq ? qb : kb) + row * 64;
  const int n = (int)(row & 8191);

  float v[64];
#pragma unroll
  for (int j = 0; j < 8; ++j) {
    uint4 raw = *(const uint4*)(p + j * 8);
    unsigned a[4] = { raw.x, raw.y, raw.z, raw.w };
#pragma unroll
    for (int t = 0; t < 4; ++t) {
      v[j * 8 + t * 2]     = bf2f((u16)(a[t] & 0xffffu));
      v[j * 8 + t * 2 + 1] = bf2f((u16)(a[t] >> 16));
    }
  }
  float s = 0.f;
#pragma unroll
  for (int d = 0; d < 64; ++d) s += v[d] * v[d];
  const float inv = 1.0f / fmaxf(sqrtf(s), 1e-12f);
  const float* sc = isq ? scq : sck;
  const float mul = isq ? (8.0f * inv) : inv;
#pragma unroll
  for (int d = 0; d < 64; ++d) v[d] *= mul * sc[d];

  const float ph = (float)(isq ? (n + 128) : n);
  float o_[64];
#pragma unroll
  for (int d = 0; d < 32; ++d) {
    const float ang = ph * invf[d];
    const float cs = __cosf(ang), sn = __sinf(ang);
    o_[d]      = v[d] * cs - v[d + 32] * sn;
    o_[d + 32] = v[d + 32] * cs + v[d] * sn;
  }
#pragma unroll
  for (int j = 0; j < 8; ++j) {
    uint4 raw;
    unsigned a[4];
#pragma unroll
    for (int t = 0; t < 4; ++t)
      a[t] = pack2(f2bf(o_[j * 8 + t * 2]), f2bf(o_[j * 8 + t * 2 + 1]));
    raw.x = a[0]; raw.y = a[1]; raw.z = a[2]; raw.w = a[3];
    *(uint4*)(p + j * 8) = raw;
  }
}

// ---------------------------------------------------------------------------
// Kernel 3: windowed attention, flash-style over 3 key chunks of 128.
// k/v tiles staged by the Tensor Data Mover (TDM) when available (k tile uses
// D# pad_enable to produce the bank-conflict-free 72-u16 row stride directly);
// falls back to per-lane async global->LDS otherwise.
// ---------------------------------------------------------------------------
__global__ __launch_bounds__(256, 1)
void attn_kernel(const u16* __restrict__ qbuf, const u16* __restrict__ kbuf,
                 const u16* __restrict__ vbuf, u16* __restrict__ attn) {
  __shared__ u16 k_lds[128 * 72];     // [key][d] padded (row = 128B data + 16B pad)
  __shared__ u16 v_lds[128 * 64];     // staging
  __shared__ u16 vt_lds[64 * 136];    // [d][key] padded
  __shared__ u16 p_lds[8 * 16 * 136]; // per-wave probs 16x128 padded

  const int tid = threadIdx.x, lane = tid & 31, wave = tid >> 5;
  const int lc = lane & 15, kg = lane >> 4;
  const int w = blockIdx.x;           // 0..63
  const int bh = blockIdx.y;          // 0..31
  const int pbase = wave * (16 * 136);

  Frag qa[2];
  {
    const int n0q = w * 128 + wave * 16 + lc;
    const u16* qr = qbuf + ((size_t)bh * 8192 + n0q) * 64;
#pragma unroll
    for (int ks = 0; ks < 2; ++ks) {
      qa[ks].u.lo = *(const uint4*)(qr + ks * 32 + kg * 8);
      qa[ks].u.hi = *(const uint4*)(qr + ks * 32 + kg * 8 + 16);
    }
  }

  f32x8 o[4] = {};
  float mrow[8], lrow[8];
#pragma unroll
  for (int r = 0; r < 8; ++r) { mrow[r] = -3.0e38f; lrow[r] = 0.f; }

  for (int c = 0; c < 3; ++c) {
    const int cw = w - 1 + c;
    if (cw < 0 || cw >= 64) continue;          // padded chunk: fully masked
    const int jabs0 = cw * 128;
    __syncthreads();                           // protect LDS reuse
    const size_t gelt = ((size_t)bh * 8192 + jabs0) * 64;   // element offset of tile
#ifdef HAVE_TDM
    if (wave == 0) {
      // D# group0: count=1 | lds_addr | 57b global_addr | type=2
      const unsigned long long gaK = (unsigned long long)(uintptr_t)(kbuf + gelt);
      const unsigned long long gaV = (unsigned long long)(uintptr_t)(vbuf + gelt);
      // D# group1 (element units, data_size=2B):
      //   tensor_dim0=64, tensor_dim1=262144, tile 64x128, dim0_stride=64
      //   k tile: pad_enable, pad_interval=4 (32 DW rows), pad_amount=3 (4 DW pad)
      const int g1_common1 = (int)(64u << 16);            // tensor_dim0[15:0]<<16
      const int g1_common3 = (int)(4u | (64u << 16));     // dim1 hi16 | tile_dim0
      uint32x4 g0k = { 1u, lds_addr(&k_lds[0]), (unsigned)gaK,
                       (unsigned)((gaK >> 32) & 0x01FFFFFFu) | 0x80000000u };
      int32x8 g1k = { (int)((1u << 16) | (1u << 20) | (4u << 22) | (3u << 25)),
                      g1_common1, 0, g1_common3, 128, 64, 0, 0 };
      uint32x4 g0v = { 1u, lds_addr(&v_lds[0]), (unsigned)gaV,
                       (unsigned)((gaV >> 32) & 0x01FFFFFFu) | 0x80000000u };
      int32x8 g1v = { (int)(1u << 16), g1_common1, 0, g1_common3, 128, 64, 0, 0 };
      int32x4 z4 = { 0, 0, 0, 0 };
      int32x8 z8 = { 0, 0, 0, 0, 0, 0, 0, 0 };
      __builtin_amdgcn_tensor_load_to_lds(g0k, g1k, z4, z4, z8, 0);
      __builtin_amdgcn_tensor_load_to_lds(g0v, g1v, z4, z4, z8, 0);
      __builtin_amdgcn_s_wait_tensorcnt(0);
    }
#else
    {
      const unsigned gbase = (unsigned)(gelt * 2);         // byte offset
#pragma unroll
      for (int j = 0; j < 4; ++j) {
        const int u = tid + j * 256;           // b128 index 0..1023
        const int row = u >> 2, pos = u & 3;
        async_b128(lds_addr(&k_lds[0]) + (unsigned)(row * 144 + pos * 16), kbuf,
                   gbase + (unsigned)(row * 128 + pos * 16));
        async_b128(lds_addr(&v_lds[0]) + (unsigned)(u * 16), vbuf,
                   gbase + (unsigned)(u * 16));
      }
    }
    wait_async_0();
#endif
    __syncthreads();
    // transpose v -> vt
    for (int j = 0; j < 32; ++j) {
      const int idx = tid + j * 256;
      const int d = idx >> 7, key = idx & 127;
      vt_lds[d * 136 + key] = v_lds[key * 64 + d];
    }
    // sim = q @ k^T  (16 x 128 per wave), masked
    f32x8 s[8];
#pragma unroll
    for (int nt = 0; nt < 8; ++nt) {
      f32x8 acc = {};
#pragma unroll
      for (int ks = 0; ks < 2; ++ks) {
        Frag b;
        const int key = nt * 16 + lc;
        const int db = ks * 32 + kg * 8;
        b.u.lo = *(const uint4*)&k_lds[key * 72 + db];
        b.u.hi = *(const uint4*)&k_lds[key * 72 + db + 16];
        acc = wmma_bf16(qa[ks].v, b.v, acc);
      }
      const int jl = c * 128 + nt * 16 + lc;   // index inside 384 look-window
#pragma unroll
      for (int r = 0; r < 8; ++r) {
        const int il = wave * 16 + r + 8 * kg; // query row in window
        const bool valid = (jl >= il) && (jl <= il + 256);
        acc[r] = valid ? acc[r] : -3.0e38f;
      }
      s[nt] = acc;
    }
    __syncthreads();                           // vt complete, k/v reads done
    // online softmax (row = component r, half-group kg; 16 lanes = 16 cols)
#pragma unroll
    for (int r = 0; r < 8; ++r) {
      float mx = -3.0e38f;
#pragma unroll
      for (int nt = 0; nt < 8; ++nt) mx = fmaxf(mx, s[nt][r]);
#pragma unroll
      for (int msk = 1; msk < 16; msk <<= 1) mx = fmaxf(mx, __shfl_xor(mx, msk, 32));
      const float mnew = fmaxf(mrow[r], mx);
      const float alpha = __expf(mrow[r] - mnew);
      float sum = 0.f;
#pragma unroll
      for (int nt = 0; nt < 8; ++nt) {
        const float pp = __expf(s[nt][r] - mnew);
        s[nt][r] = pp; sum += pp;
      }
#pragma unroll
      for (int msk = 1; msk < 16; msk <<= 1) sum += __shfl_xor(sum, msk, 32);
      lrow[r] = lrow[r] * alpha + sum;
      mrow[r] = mnew;
#pragma unroll
      for (int ont = 0; ont < 4; ++ont) o[ont][r] *= alpha;
    }
    // probs (C-layout) -> per-wave LDS (row-major) for A-fragments
#pragma unroll
    for (int nt = 0; nt < 8; ++nt)
#pragma unroll
      for (int r = 0; r < 8; ++r)
        p_lds[pbase + (r + 8 * kg) * 136 + nt * 16 + lc] = f2bf(s[nt][r]);
    // out += probs @ v   (K = 128 in 4 steps, N = 64 in 4 tiles)
#pragma unroll
    for (int ks4 = 0; ks4 < 4; ++ks4) {
      Frag pa;
      const int kkk = ks4 * 32 + kg * 8;
      pa.u.lo = *(const uint4*)&p_lds[pbase + lc * 136 + kkk];
      pa.u.hi = *(const uint4*)&p_lds[pbase + lc * 136 + kkk + 16];
#pragma unroll
      for (int ont = 0; ont < 4; ++ont) {
        Frag vbf;
        const int d = ont * 16 + lc;
        const int kb2 = ks4 * 32 + kg * 8;
        vbf.u.lo = *(const uint4*)&vt_lds[d * 136 + kb2];
        vbf.u.hi = *(const uint4*)&vt_lds[d * 136 + kb2 + 16];
        o[ont] = wmma_bf16(pa.v, vbf.v, o[ont]);
      }
    }
  }

  // epilogue: normalize and write bf16 to [b*8192+n][h*64+d]
  const int bb = bh >> 4, hh = bh & 15;
#pragma unroll
  for (int ont = 0; ont < 4; ++ont)
#pragma unroll
    for (int r = 0; r < 8; ++r) {
      const int row = wave * 16 + r + 8 * kg;
      const int nabs = w * 128 + row;
      const int col = hh * 64 + ont * 16 + lc;
      attn[((size_t)(bb * 8192 + nabs)) * 1024 + col] = f2bf(o[ont][r] / lrow[r]);
    }
}

// ---------------------------------------------------------------------------
// Kernel 4: y = attn(16384x1024 bf16) @ w_out^T(1024x1024 bf16) -> f32
// ---------------------------------------------------------------------------
__global__ __launch_bounds__(256, 1)
void out_gemm(const u16* __restrict__ a_g, const u16* __restrict__ woT,
              float* __restrict__ out) {
  __shared__ u16 a_lds[2][128 * 40];
  __shared__ u16 b_lds[2][128 * 40];
  const int tid = threadIdx.x, lane = tid & 31, wave = tid >> 5;
  const int lc = lane & 15, kg = lane >> 4;
  const int m0 = blockIdx.y * 128, n0 = blockIdx.x * 128;
  const int wm = wave >> 1, wn = wave & 1;
  f32x8 acc[2][4] = {};

  auto issue = [&](int buf, int kk) {
#pragma unroll
    for (int j = 0; j < 2; ++j) {
      const int idx = tid + j * 256;
      const int row = idx >> 2, pos = idx & 3;
      const unsigned l = (unsigned)((row * 40 + pos * 8) * 2);
      async_b128(lds_addr(&a_lds[buf][0]) + l, a_g,
                 (unsigned)(((m0 + row) * 1024 + kk + pos * 8) * 2));
      async_b128(lds_addr(&b_lds[buf][0]) + l, woT,
                 (unsigned)(((n0 + row) * 1024 + kk + pos * 8) * 2));
    }
  };

  issue(0, 0);
  for (int it = 0; it < 32; ++it) {
    const int buf = it & 1;
    if (it + 1 < 32) { issue(buf ^ 1, (it + 1) * 32); wait_async_le4(); }
    else             { wait_async_0(); }
    __syncthreads();

    Frag a[2], b[4];
#pragma unroll
    for (int mt = 0; mt < 2; ++mt) {
      const int m = wm * 32 + mt * 16 + lc;
      a[mt].u.lo = *(const uint4*)&a_lds[buf][m * 40 + kg * 8];
      a[mt].u.hi = *(const uint4*)&a_lds[buf][m * 40 + kg * 8 + 16];
    }
#pragma unroll
    for (int nt = 0; nt < 4; ++nt) {
      const int nn = wn * 64 + nt * 16 + lc;
      b[nt].u.lo = *(const uint4*)&b_lds[buf][nn * 40 + kg * 8];
      b[nt].u.hi = *(const uint4*)&b_lds[buf][nn * 40 + kg * 8 + 16];
    }
#pragma unroll
    for (int mt = 0; mt < 2; ++mt)
#pragma unroll
      for (int nt = 0; nt < 4; ++nt)
        acc[mt][nt] = wmma_bf16(a[mt].v, b[nt].v, acc[mt][nt]);
    __syncthreads();
  }

#pragma unroll
  for (int mt = 0; mt < 2; ++mt)
#pragma unroll
    for (int nt = 0; nt < 4; ++nt)
#pragma unroll
      for (int r = 0; r < 8; ++r) {
        const int row = m0 + wm * 32 + mt * 16 + r + 8 * kg;
        const int col = n0 + wn * 64 + nt * 16 + lc;
        out[(size_t)row * 1024 + col] = acc[mt][nt][r];
      }
}

// ---------------------------------------------------------------------------
extern "C" void kernel_launch(void* const* d_in, const int* in_sizes, int n_in,
                              void* d_out, int out_size, void* d_ws, size_t ws_size,
                              hipStream_t stream) {
  const float* x    = (const float*)d_in[0];
  const float* wqkv = (const float*)d_in[1];
  const float* wout = (const float*)d_in[2];
  const float* qs   = (const float*)d_in[3];
  const float* ksc  = (const float*)d_in[4];
  float* out = (float*)d_out;

  char* ws = (char*)d_ws;
  u16* xb  = (u16*)(ws);                      // 16384*1024 bf16 = 33,554,432 B
  u16* wqT = (u16*)(ws + 33554432);           // 3072*1024 bf16  =  6,291,456 B
  u16* woT = (u16*)(ws + 39845888);           // 1024*1024 bf16  =  2,097,152 B
  u16* qb  = (u16*)(ws + 41943040);           // 32*8192*64 bf16 = 33,554,432 B
  u16* kb  = (u16*)(ws + 75497472);
  u16* vb  = (u16*)(ws + 109051904);
  u16* ab  = (u16*)(ws + 142606336);          // attn out, GEMM-ready layout

  prep_kernel<<<81920, 256, 0, stream>>>(x, wqkv, wout, xb, wqT, woT);
  qkv_gemm<<<dim3(24, 128), 256, 0, stream>>>(xb, wqT, qb, kb, vb);
  normrope_kernel<<<2048, 256, 0, stream>>>(qb, kb, qs, ksc);
  attn_kernel<<<dim3(64, 32), 256, 0, stream>>>(qb, kb, vb, ab);
  out_gemm<<<dim3(8, 128), 256, 0, stream>>>(ab, woT, out);
}
